// GatedGraphConvolution_21337397526884
// MI455X (gfx1250) — compile-verified, matrix operand order checked by reading, not run on hardware
//
#include <hip/hip_runtime.h>

// GatedGraphConvolution for MI455X (gfx1250, wave32, WMMA).
// Pipeline: [GEMM x3 fused epilogues, bf16 WMMA f32-acc] -> [edge scatter-add] -> [gated combine]

#define NDIM 128
#define LDS_PAD 4
#define XROW (NDIM + LDS_PAD)   // bf16 elements per LDS row (keeps u32 alignment, spreads banks)

typedef __bf16 bf16;
typedef __attribute__((ext_vector_type(16))) __bf16 v16bf;
typedef __attribute__((ext_vector_type(8)))  float  v8f;

union FragAB { v16bf v; unsigned u[8]; };

// ---------------------------------------------------------------------------
// Kernel 1: one block computes a 128-row x 128-col tile of x @ W[which].
//   which==0: S = support ; A = eps*support + b1   (agg accumulator init)
//   which==1: T = x@w2 + b2
//   which==2: G = sigmoid(x@w3 + b3)
// 8 waves per block, each wave owns a 16x128 slab (8 accumulators of 16x16).
// ---------------------------------------------------------------------------
__global__ __launch_bounds__(256) void ggc_gemm(
    const float* __restrict__ x,
    const float* __restrict__ w1, const float* __restrict__ w2, const float* __restrict__ w3,
    const float* __restrict__ b1, const float* __restrict__ b2, const float* __restrict__ b3,
    const float* __restrict__ epsA,
    float* __restrict__ S, float* __restrict__ A,
    float* __restrict__ T, float* __restrict__ G,
    int nNodes)
{
    __shared__ bf16 sX[128 * XROW];  // x tile, row-major [row][k], bf16
    __shared__ bf16 sW[128 * XROW];  // W tile, transposed [n][k], bf16

    const int t     = threadIdx.x;
    const int wave  = t >> 5;
    const int lane  = t & 31;
    const int which = blockIdx.y;
    const int row0  = blockIdx.x * 128;
    const float* W  = (which == 0) ? w1 : (which == 1) ? w2 : w3;

    // ---- Stage x tile (f32 -> bf16), coalesced float4 loads ----
    #pragma unroll
    for (int it = 0; it < 16; ++it) {
        int idx4 = t + it * 256;          // 4096 float4 = 128x128
        int r    = idx4 >> 5;
        int c    = (idx4 & 31) * 4;
        int grow = row0 + r;
        float4 v = make_float4(0.f, 0.f, 0.f, 0.f);
        if (grow < nNodes) v = *(const float4*)(x + (size_t)grow * NDIM + c);
        bf16* p = &sX[r * XROW + c];
        p[0] = (bf16)v.x; p[1] = (bf16)v.y; p[2] = (bf16)v.z; p[3] = (bf16)v.w;
    }
    // ---- Stage W transposed: sW[n][k] so B-fragment k-pairs are contiguous ----
    #pragma unroll
    for (int it = 0; it < 16; ++it) {
        int idx4 = t + it * 256;
        int k    = idx4 >> 5;
        int n    = (idx4 & 31) * 4;
        float4 v = *(const float4*)(W + k * NDIM + n);
        sW[(n + 0) * XROW + k] = (bf16)v.x;
        sW[(n + 1) * XROW + k] = (bf16)v.y;
        sW[(n + 2) * XROW + k] = (bf16)v.z;
        sW[(n + 3) * XROW + k] = (bf16)v.w;
    }
    __syncthreads();

    const int m    = lane & 15;
    const int half = lane >> 4;
    const int arow = wave * 16 + m;

    v8f acc[8] = {};   // 8 N-tiles of 16x16 f32

    // K = 128 in 4 chunks of 32 (one wmma_f32_16x16x32_bf16 per N-tile per chunk)
    #pragma unroll
    for (int kc = 0; kc < 4; ++kc) {
        // A fragment per ISA layout: lane(m,half); VGPR v<4: K=K0+half*8+2v, v>=4: +16
        FragAB a;
        const unsigned* xr = (const unsigned*)&sX[arow * XROW];
        const int base = kc * 16 + half * 4;   // u32 index (2 bf16 per u32)
        #pragma unroll
        for (int vv = 0; vv < 4; ++vv) { a.u[vv] = xr[base + vv]; a.u[4 + vv] = xr[base + 8 + vv]; }

        #pragma unroll
        for (int nt = 0; nt < 8; ++nt) {
            FragAB b;  // B fragment: lane -> column (nt*16+m), same K packing as A
            const unsigned* wr = (const unsigned*)&sW[(nt * 16 + m) * XROW];
            #pragma unroll
            for (int vv = 0; vv < 4; ++vv) { b.u[vv] = wr[base + vv]; b.u[4 + vv] = wr[base + 8 + vv]; }

            acc[nt] = __builtin_amdgcn_wmma_f32_16x16x32_bf16(
                false, a.v, false, b.v, (short)0, acc[nt], false, false);
        }
    }

    // ---- Fused epilogues; C/D layout: VGPR r, lanes16-31 => M=r+8 ----
    const float eps = epsA[0];
    #pragma unroll
    for (int nt = 0; nt < 8; ++nt) {
        const int col = nt * 16 + m;
        #pragma unroll
        for (int r = 0; r < 8; ++r) {
            const int row = row0 + wave * 16 + r + half * 8;
            if (row < nNodes) {
                const size_t o = (size_t)row * NDIM + col;
                const float c = acc[nt][r];
                if (which == 0) {
                    S[o] = c;                       // support (gathered by SpMM)
                    A[o] = eps * c + b1[col];       // accumulator init: agg += edges later
                } else if (which == 1) {
                    T[o] = c + b2[col];             // trans + b2
                } else {
                    const float g = c + b3[col];
                    G[o] = 1.0f / (1.0f + __expf(-g));  // sigmoid(gate + b3)
                }
            }
        }
    }
}

// ---------------------------------------------------------------------------
// Kernel 2: COO SpMM scatter-add. 32 lanes per edge, float4 per lane.
// A[dst] += w_e * S[src]. S and A are L2-resident (51.2 MB each < 192 MB L2).
// ---------------------------------------------------------------------------
__global__ __launch_bounds__(256) void ggc_spmm(
    const float* __restrict__ S,
    const int*   __restrict__ esrc,
    const int*   __restrict__ edst,
    const float* __restrict__ ew,
    float* __restrict__ A,
    int nEdges)
{
    long tid = (long)blockIdx.x * blockDim.x + threadIdx.x;
    int e = (int)(tid >> 5);
    if (e >= nEdges) return;
    int d = ((int)tid & 31) * 4;

    const int   s = esrc[e];
    const int   q = edst[e];
    const float w = ew[e];

    const float4 sv = *(const float4*)(S + (size_t)s * NDIM + d);
    float* ap = A + (size_t)q * NDIM + d;
    atomicAdd(ap + 0, w * sv.x);
    atomicAdd(ap + 1, w * sv.y);
    atomicAdd(ap + 2, w * sv.z);
    atomicAdd(ap + 3, w * sv.w);
}

// ---------------------------------------------------------------------------
// Kernel 3: out = T + G * (relu(A) - T)   (A already holds agg + eps*S + b1)
// ---------------------------------------------------------------------------
__global__ __launch_bounds__(256) void ggc_final(
    const float* __restrict__ A,
    const float* __restrict__ T,
    const float* __restrict__ G,
    float* __restrict__ out,
    long n4)
{
    long tid = (long)blockIdx.x * blockDim.x + threadIdx.x;
    if (tid >= n4) return;
    const size_t o = (size_t)tid * 4;
    const float4 a = *(const float4*)(A + o);
    const float4 t = *(const float4*)(T + o);
    const float4 g = *(const float4*)(G + o);
    float4 r;
    r.x = t.x + g.x * (fmaxf(a.x, 0.f) - t.x);
    r.y = t.y + g.y * (fmaxf(a.y, 0.f) - t.y);
    r.z = t.z + g.z * (fmaxf(a.z, 0.f) - t.z);
    r.w = t.w + g.w * (fmaxf(a.w, 0.f) - t.w);
    *(float4*)(out + o) = r;
}

// ---------------------------------------------------------------------------
extern "C" void kernel_launch(void* const* d_in, const int* in_sizes, int n_in,
                              void* d_out, int out_size, void* d_ws, size_t ws_size,
                              hipStream_t stream)
{
    const float* x    = (const float*)d_in[0];
    const int*   esrc = (const int*)  d_in[1];
    const int*   edst = (const int*)  d_in[2];
    const float* ew   = (const float*)d_in[3];
    const float* w1   = (const float*)d_in[4];
    const float* w2   = (const float*)d_in[5];
    const float* w3   = (const float*)d_in[6];
    const float* b1   = (const float*)d_in[7];
    const float* b2   = (const float*)d_in[8];
    const float* b3   = (const float*)d_in[9];
    const float* eps  = (const float*)d_in[10];

    const int nNodes = in_sizes[0] / NDIM;
    const int nEdges = in_sizes[1];

    float* ws = (float*)d_ws;
    const size_t per = (size_t)nNodes * NDIM;
    float* S = ws + 0 * per;   // support
    float* A = ws + 1 * per;   // agg accumulator (init = eps*S + b1)
    float* T = ws + 2 * per;   // trans + b2
    float* G = ws + 3 * per;   // sigmoid(gate + b3)

    // 1) three projections with fused epilogues
    dim3 g1((unsigned)((nNodes + 127) / 128), 3, 1);
    ggc_gemm<<<g1, 256, 0, stream>>>(x, w1, w2, w3, b1, b2, b3, eps,
                                     S, A, T, G, nNodes);

    // 2) edge scatter-add into A
    long spmmThreads = (long)nEdges * 32;
    int  g2 = (int)((spmmThreads + 255) / 256);
    ggc_spmm<<<g2, 256, 0, stream>>>(S, esrc, edst, ew, A, nEdges);

    // 3) gated combine -> d_out
    long n4 = (long)nNodes * (NDIM / 4);
    int  g3 = (int)((n4 + 255) / 256);
    ggc_final<<<g3, 256, 0, stream>>>(A, T, G, (float*)d_out, n4);
}
